// ScaledDotProductAttention_62259845923351
// MI455X (gfx1250) — matrix operand order repaired and checked by reading, not compiled, not run on hardware
//
#include <hip/hip_runtime.h>

typedef __attribute__((ext_vector_type(16))) _Float16 v16h;
typedef __attribute__((ext_vector_type(8)))  _Float16 v8h;
typedef __attribute__((ext_vector_type(8)))  float    v8f;

#define B_ 4
#define H_ 16
#define S_ 2048
#define D_ 64
// 0.125 (=1/sqrt(64)) * log2(e): folded into Q at conversion time.
#define SCL 0.18033688011112042f

// Async global->LDS DMA path (CDNA5). Flip to 0 for register-copy fallback.
#define USE_ASYNC_LDS 1

static __device__ inline void copy16_to_lds(const _Float16* g, _Float16* l) {
#if USE_ASYNC_LDS
  unsigned lds_off = (unsigned)(unsigned long long)l;  // low 32b = LDS offset
  unsigned long long ga = (unsigned long long)g;
  asm volatile("global_load_async_to_lds_b128 %0, %1, off"
               :: "v"(lds_off), "v"(ga) : "memory");
#else
  *(v8h*)l = *(const v8h*)g;
#endif
}

// Wait until at most N async ops remain outstanding (they retire in order).
static __device__ inline void async_wait_le6() {
#if USE_ASYNC_LDS
  asm volatile("s_wait_asynccnt 0x6" ::: "memory");
#endif
}
static __device__ inline void async_wait_le0() {
#if USE_ASYNC_LDS
  asm volatile("s_wait_asynccnt 0x0" ::: "memory");
#endif
}

static __device__ inline v8f wmma_f16(v16h a, v16h b, v8f c) {
  return __builtin_amdgcn_wmma_f32_16x16x32_f16(false, a, false, b, (short)0, c,
                                                false, false);
}

static __device__ inline v16h a_frag_from_pairs(const _Float16* p0,
                                                const _Float16* p1) {
  v8h lo = *(const v8h*)p0;
  v8h hi = *(const v8h*)p1;
  v16h r;
#pragma unroll
  for (int i = 0; i < 8; ++i) { r[i] = lo[i]; r[8 + i] = hi[i]; }
  return r;
}

// XOR lane swap via ds_swizzle (immediate pattern, no address VALU).
template <int MASK>
static __device__ inline float swz_xor(float v) {
  int i = __builtin_amdgcn_ds_swizzle(__float_as_int(v), (MASK << 10) | 0x1f);
  return __int_as_float(i);
}
static __device__ inline float redmax16(float v) {
  v = fmaxf(v, swz_xor<1>(v));
  v = fmaxf(v, swz_xor<2>(v));
  v = fmaxf(v, swz_xor<4>(v));
  v = fmaxf(v, swz_xor<8>(v));
  return v;
}

// ---------------- pre-pass: fp32 -> fp16; Q pre-scaled; V transposed ---------
// Vt layout: [bh][d][S], keys interleaved within each 32-block (slot f holds
// key f/2 if f even else 16+f/2) to match P's packed store order.
__global__ __launch_bounds__(256)
void cvt_kernel(const float* __restrict__ Q, const float* __restrict__ K,
                const float* __restrict__ V, _Float16* __restrict__ Qh,
                _Float16* __restrict__ Kh, _Float16* __restrict__ Vt) {
  const int i = blockIdx.x * 256 + threadIdx.x;  // float2 index
  const float2* q2 = (const float2*)Q;
  const float2* k2 = (const float2*)K;
  const float2* v2 = (const float2*)V;
  float2 f = q2[i];
  {
    auto pk = __builtin_amdgcn_cvt_pkrtz(f.x * SCL, f.y * SCL);
    *(decltype(pk)*)(Qh + 2 * i) = pk;
  }
  f = k2[i];
  {
    auto pk = __builtin_amdgcn_cvt_pkrtz(f.x, f.y);
    *(decltype(pk)*)(Kh + 2 * i) = pk;
  }
  f = v2[i];
  const int e   = 2 * i;
  const int bhv = e >> 17;           // / (S*D)
  const int rem = e & (S_ * D_ - 1);
  const int sv  = rem >> 6;          // sequence index
  const int dv  = rem & (D_ - 1);    // d, d+1
  const int kk  = sv & 31;
  const int fpos = (kk < 16) ? (2 * kk) : (2 * (kk - 16) + 1);
  const int svp  = (sv & ~31) | fpos;
  Vt[((size_t)bhv * D_ + dv) * S_ + svp]     = (_Float16)f.x;
  Vt[((size_t)bhv * D_ + dv + 1) * S_ + svp] = (_Float16)f.y;
}

// ---------------- main flash-attention kernel --------------------------------
// 4 waves/block, each wave owns 32 q rows (two 16-row subtiles) -> 128 q/block.
// Double-buffered async K/V staging: DMA for block i+1 overlaps compute of i.
__global__ __launch_bounds__(128)
void fa_causal_wmma_kernel(const _Float16* __restrict__ Qh,
                           const _Float16* __restrict__ Kh,
                           const _Float16* __restrict__ Vt,
                           float* __restrict__ O) {
  __shared__ __attribute__((aligned(32))) _Float16 sK[2][32 * 64];   // [key][d]
  __shared__ __attribute__((aligned(32))) _Float16 sVt[2][64 * 32];  // [d][slot]
  __shared__ __attribute__((aligned(32))) _Float16 sP[4][16 * 32];

  const int tid  = threadIdx.x;
  const int wave = tid >> 5;
  const int lane = tid & 31;
  const int lh   = lane & 15;
  const int grp  = lane >> 4;
  const int sel8 = grp * 8;

  const int bh    = blockIdx.y;
  const size_t bo = (size_t)bh * S_ * D_;
  const int qbase = blockIdx.x * 128;
  const int q0    = qbase + wave * 32;

  v16h aQ[2][2];
#pragma unroll
  for (int t = 0; t < 2; ++t) {
    const _Float16* qrow = Qh + bo + (size_t)(q0 + t * 16 + lh) * D_;
    aQ[t][0] = a_frag_from_pairs(qrow + 2 * sel8, qrow + 8 + 2 * sel8);
    aQ[t][1] = a_frag_from_pairs(qrow + 32 + 2 * sel8, qrow + 40 + 2 * sel8);
  }

  v16h bOnes;
#pragma unroll
  for (int i = 0; i < 16; ++i) bOnes[i] = (_Float16)1.0f;

  v8f o[2][4] = {};
  float m[2][8], l[2][8];
#pragma unroll
  for (int t = 0; t < 2; ++t)
#pragma unroll
    for (int r = 0; r < 8; ++r) { m[t][r] = -__builtin_inff(); l[t][r] = 0.f; }

  const int jend = qbase + 128;

  // Issue 4 async 16B-chunk copies staging K (32x64) and Vt (64x32) tiles.
  auto issue_tiles = [&](int buf, int jb) {
    const _Float16* kg = Kh + bo + (size_t)jb * D_;
    copy16_to_lds(kg + tid * 8, sK[buf] + tid * 8);
    copy16_to_lds(kg + (tid + 128) * 8, sK[buf] + (tid + 128) * 8);
    const _Float16* vg = Vt + (size_t)bh * D_ * S_ + jb;
    const int c0 = tid, c1 = tid + 128;
    copy16_to_lds(vg + (size_t)(c0 >> 2) * S_ + (c0 & 3) * 8,
                  sVt[buf] + (c0 >> 2) * 32 + (c0 & 3) * 8);
    copy16_to_lds(vg + (size_t)(c1 >> 2) * S_ + (c1 & 3) * 8,
                  sVt[buf] + (c1 >> 2) * 32 + (c1 & 3) * 8);
  };

  int ibuf = 0;
  issue_tiles(0, 0);  // prologue: stage first tile pair

  for (int jb = 0; jb < jend; jb += 32) {
    const bool more = (jb + 32) < jend;
    __syncthreads();  // nobody still reads the buffer we are about to fill
    if (more) {
      issue_tiles(ibuf ^ 1, jb + 32);  // overlap DMA with this block's compute
      async_wait_le6();                // <=6 of 8 pending; current tiles are the older 4 ops
    } else {
      async_wait_le0();
    }
    __syncthreads();  // publish completed tiles to all waves

    if (jb < q0 + 32) {  // wave-uniform
      const _Float16* cK = sK[ibuf];
      const _Float16* cV = sVt[ibuf];
      v16h bK00 = *(const v16h*)(cK + (0 + lh) * 64 + 0 + 16 * grp);
      v16h bK01 = *(const v16h*)(cK + (0 + lh) * 64 + 32 + 16 * grp);
      v16h bK10 = *(const v16h*)(cK + (16 + lh) * 64 + 0 + 16 * grp);
      v16h bK11 = *(const v16h*)(cK + (16 + lh) * 64 + 32 + 16 * grp);

#pragma unroll
      for (int t = 0; t < 2; ++t) {
        const int q0t = q0 + t * 16;
        if (jb >= q0t + 16) continue;  // wave-uniform skip

        v8f s0 = {}, s1 = {};
        s0 = wmma_f16(aQ[t][0], bK00, s0);
        s0 = wmma_f16(aQ[t][1], bK01, s0);
        s1 = wmma_f16(aQ[t][0], bK10, s1);
        s1 = wmma_f16(aQ[t][1], bK11, s1);

        _Float16* pw = &sP[wave][0];

        auto softmax_pass = [&](int masked) {
#pragma unroll
          for (int r = 0; r < 8; ++r) {
            float x0 = s0[r];  // scale pre-folded into Q
            float x1 = s1[r];
            if (masked) {
              const int q = q0t + r + 8 * grp;
              if (jb + lh > q) x0 = -__builtin_inff();
              if (jb + 16 + lh > q) x1 = -__builtin_inff();
            }
            const float mnew = fmaxf(m[t][r], redmax16(fmaxf(x0, x1)));
            const float alpha = __builtin_amdgcn_exp2f(m[t][r] - mnew);
            m[t][r] = mnew;
            const float p0 = __builtin_amdgcn_exp2f(x0 - mnew);
            const float p1 = __builtin_amdgcn_exp2f(x1 - mnew);
            l[t][r] *= alpha;  // row-sum of new P added via WMMA below
            o[t][0][r] *= alpha; o[t][1][r] *= alpha;
            o[t][2][r] *= alpha; o[t][3][r] *= alpha;
            auto pk = __builtin_amdgcn_cvt_pkrtz(p0, p1);
            *(decltype(pk)*)(pw + (r + 8 * grp) * 32 + 2 * lh) = pk;
          }
        };
        if (jb + 31 > q0t) softmax_pass(1); else softmax_pass(0);

        v16h aP = a_frag_from_pairs(pw + lh * 32 + 2 * sel8,
                                    pw + lh * 32 + 8 + 2 * sel8);

        v8f zero = {};
        v8f ls = wmma_f16(aP, bOnes, zero);  // row-sum(P), replicated over N
#pragma unroll
        for (int r = 0; r < 8; ++r) l[t][r] += ls[r];

        v16h bV;
        bV = *(const v16h*)(cV + (0 + lh) * 32 + 16 * grp);
        o[t][0] = wmma_f16(aP, bV, o[t][0]);
        bV = *(const v16h*)(cV + (16 + lh) * 32 + 16 * grp);
        o[t][1] = wmma_f16(aP, bV, o[t][1]);
        bV = *(const v16h*)(cV + (32 + lh) * 32 + 16 * grp);
        o[t][2] = wmma_f16(aP, bV, o[t][2]);
        bV = *(const v16h*)(cV + (48 + lh) * 32 + 16 * grp);
        o[t][3] = wmma_f16(aP, bV, o[t][3]);
      }
    }
    ibuf ^= 1;
  }

#pragma unroll
  for (int t = 0; t < 2; ++t)
#pragma unroll
    for (int r = 0; r < 8; ++r) {
      const float inv = 1.0f / l[t][r];
      const int q = q0 + t * 16 + r + 8 * grp;
      float* op = O + bo + (size_t)q * D_ + lh;
      op[0]  = o[t][0][r] * inv;
      op[16] = o[t][1][r] * inv;
      op[32] = o[t][2][r] * inv;
      op[48] = o[t][3][r] * inv;
    }
}

extern "C" void kernel_launch(void* const* d_in, const int* in_sizes, int n_in,
                              void* d_out, int out_size, void* d_ws,
                              size_t ws_size, hipStream_t stream) {
  (void)in_sizes; (void)n_in; (void)out_size; (void)ws_size;
  const float* Q = (const float*)d_in[0];
  const float* K = (const float*)d_in[1];
  const float* V = (const float*)d_in[2];
  float* Out = (float*)d_out;  // d_in[3] (mask) ignored: causal is analytic

  const size_t n = (size_t)B_ * H_ * S_ * D_;
  _Float16* Qh = (_Float16*)d_ws;
  _Float16* Kh = Qh + n;
  _Float16* Vt = Kh + n;  // needs 3*n*2 = 50.3 MB workspace

  cvt_kernel<<<(unsigned)(n / 2 / 256), 256, 0, stream>>>(Q, K, V, Qh, Kh, Vt);
  dim3 grid(S_ / 128, B_ * H_);
  fa_causal_wmma_kernel<<<grid, 128, 0, stream>>>(Qh, Kh, Vt, Out);
}
